// ContinuousThoughtMachine_79293686218892
// MI455X (gfx1250) — matrix operand order repaired and checked by reading, not compiled, not run on hardware
//
#include <hip/hip_runtime.h>
#include <hip/hip_bf16.h>

// ---------------------------------------------------------------------------
// CDNA5 / gfx1250 ContinuousThoughtMachine forward.
//  - all GEMMs on v_wmma_f32_16x16x32_bf16 (f32 accumulate)
//  - weights/activations feeding WMMA pre-converted to bf16 (halves traffic;
//    enables direct global->LDS async staging)
//  - tile staging via GLOBAL_LOAD_ASYNC_TO_LDS_B128 (ASYNCcnt) when the
//    builtin exists; synchronous b128 copy fallback otherwise
//  - K/V kept bf16 (134 MB) so both stay resident in the 192 MB L2 across the
//    16-step attention sweep
// ---------------------------------------------------------------------------

#define DEV static __device__ __forceinline__

typedef __attribute__((ext_vector_type(16))) __bf16 bf16x16;
typedef __attribute__((ext_vector_type(8)))  float  f32x8;
typedef int v4i __attribute__((vector_size(16)));   // matches builtin param type

struct Chunk2 { uint4 a, b; };   // 32 bytes == one v16bf fragment

#if defined(__HIP_DEVICE_COMPILE__) && \
    __has_builtin(__builtin_amdgcn_global_load_async_to_lds_b128)
#define USE_ASYNC_LDS 1
#else
#define USE_ASYNC_LDS 0
#endif

DEV unsigned short f2bf(float f) {               // RNE f32 -> bf16
  unsigned int u = __float_as_uint(f);
  u += 0x7FFFu + ((u >> 16) & 1u);
  return (unsigned short)(u >> 16);
}
DEV float bf2f(unsigned short h) { return __uint_as_float(((unsigned int)h) << 16); }
DEV float gelu_f(float x) { return 0.5f * x * (1.0f + erff(x * 0.70710678118654752f)); }

static constexpr int LSTR = 40;  // LDS row stride (bf16): 80B, conflict-free b128 frags

// ---- async (or fallback) 16-byte global -> LDS copy -----------------------
DEV void copy16_g2l(const unsigned short* g, unsigned short* l) {
#if USE_ASYNC_LDS
  auto gp = (__attribute__((address_space(1))) v4i*)(uintptr_t)g;
  auto lp = (__attribute__((address_space(3))) v4i*)(unsigned int)(uintptr_t)l;
  __builtin_amdgcn_global_load_async_to_lds_b128(gp, lp, 0, 0);
#else
  *reinterpret_cast<uint4*>(l) = *reinterpret_cast<const uint4*>(g);
#endif
}
DEV void wait_async_lds() {
#if USE_ASYNC_LDS
#if __has_builtin(__builtin_amdgcn_s_wait_asynccnt)
  __builtin_amdgcn_s_wait_asynccnt(0);
#else
  asm volatile("s_wait_asynccnt 0x0" ::: "memory");
#endif
#endif
}

// ---- stage a 64(row) x 32(K) bf16 tile into LDS; 256 threads, 8 K each ----
DEV void stage_tile(const unsigned short* __restrict__ src, int ld, int rowBase, int k0,
                    unsigned short* lds) {
  const int tr = threadIdx.x >> 2;          // 0..63 tile row
  const int tk = (threadIdx.x & 3) << 3;    // 0,8,16,24
  copy16_g2l(src + (size_t)(rowBase + tr) * ld + k0 + tk, &lds[tr * LSTR + tk]);
}

// A-fragment (16x32 bf16, ISA 7.12.2): lane(m=l&15) half=l>>4:
//   elems 0..7  -> K = half*8 + 0..7 ; elems 8..15 -> K = 16 + half*8 + 0..7
DEV bf16x16 frag_a(const unsigned short* lds, int row, int half) {
  Chunk2 c;
  c.a = *reinterpret_cast<const uint4*>(lds + row * LSTR + half * 8);
  c.b = *reinterpret_cast<const uint4*>(lds + row * LSTR + 16 + half * 8);
  return __builtin_bit_cast(bf16x16, c);
}
// B-fragment (32x16 bf16): lane(n=l&15) half=l>>4:
//   elems 0..7  -> K = half*8 + 0..7 ; elems 8..15 -> K = 16 + half*8 + 0..7
DEV bf16x16 frag_b(const unsigned short* lds, int row, int half) {
  Chunk2 c;
  c.a = *reinterpret_cast<const uint4*>(lds + row * LSTR + half * 8);
  c.b = *reinterpret_cast<const uint4*>(lds + row * LSTR + 16 + half * 8);
  return __builtin_bit_cast(bf16x16, c);
}

DEV void store_out(float* C, size_t idx, float v) { C[idx] = v; }
DEV void store_out(unsigned short* C, size_t idx, float v) { C[idx] = f2bf(v); }

// ---------------------------------------------------------------------------
// C = act(A @ W^T + bias).  A: (M,K) bf16, W: (N,K) bf16 row-major.
// Block tile 64x64, BK=32, 8 waves; each wave -> 16x32 via two 16x16 WMMAs.
// grid = (N/64, M/64). All M,N,K used are multiples of the tile.
// ---------------------------------------------------------------------------
template<typename TOUT, bool GELU>
__global__ __launch_bounds__(256) void gemm_wmma_kernel(
    const unsigned short* __restrict__ A, int lda,
    const unsigned short* __restrict__ W, int ldb,
    const float* __restrict__ bias,
    TOUT* __restrict__ C, int ldc, int K)
{
  __shared__ __align__(16) unsigned short lsA[64 * LSTR];
  __shared__ __align__(16) unsigned short lsB[64 * LSTR];
  const int rowBase = blockIdx.y * 64;
  const int nBase   = blockIdx.x * 64;
  const int wave = threadIdx.x >> 5, lane = threadIdx.x & 31;
  const int wr = wave >> 1, wc = wave & 1;   // 4x2 wave grid over 64x64
  const int half = lane >> 4, lrow = lane & 15;

  f32x8 c0 = {};
  f32x8 c1 = {};
  for (int k0 = 0; k0 < K; k0 += 32) {
    stage_tile(A, lda, rowBase, k0, lsA);
    stage_tile(W, ldb, nBase,  k0, lsB);
    wait_async_lds();
    __syncthreads();
    bf16x16 a  = frag_a(lsA, wr * 16 + lrow, half);
    bf16x16 b0 = frag_b(lsB, wc * 32 + lrow, half);
    bf16x16 b1 = frag_b(lsB, wc * 32 + 16 + lrow, half);
    c0 = __builtin_amdgcn_wmma_f32_16x16x32_bf16(false, a, false, b0, (short)0, c0, false, false);
    c1 = __builtin_amdgcn_wmma_f32_16x16x32_bf16(false, a, false, b1, (short)0, c1, false, false);
    __syncthreads();
  }
  // D layout: VGPR r -> row r (lanes 0-15) / r+8 (lanes 16-31); col = lane&15
  #pragma unroll
  for (int tile = 0; tile < 2; ++tile) {
    const f32x8 acc = tile ? c1 : c0;
    const int col = nBase + wc * 32 + tile * 16 + lrow;
    const float bv = bias[col];
    #pragma unroll
    for (int r = 0; r < 8; ++r) {
      const int row = rowBase + wr * 16 + half * 8 + r;
      float v = acc[r] + bv;
      if (GELU) v = gelu_f(v);
      store_out(C, (size_t)row * ldc + col, v);
    }
  }
}

// ---- f32 -> bf16 bulk conversion (n % 4 == 0) -----------------------------
__global__ void cvt_kernel(const float* __restrict__ src, unsigned short* __restrict__ dst,
                           int n4)
{
  const int i = blockIdx.x * blockDim.x + threadIdx.x;
  if (i < n4) {
    const float4 v = reinterpret_cast<const float4*>(src)[i];
    uint2 p;
    p.x = (unsigned)f2bf(v.x) | ((unsigned)f2bf(v.y) << 16);
    p.y = (unsigned)f2bf(v.z) | ((unsigned)f2bf(v.w) << 16);
    reinterpret_cast<uint2*>(dst)[i] = p;
  }
}

// ---------------------------------------------------------------------------
// LayerNorm(D=1024) + GELU, one block per row, output bf16 ctx.
// ---------------------------------------------------------------------------
__global__ __launch_bounds__(256) void ln_gelu_kernel(
    const float* __restrict__ h, const float* __restrict__ g,
    const float* __restrict__ bvec, unsigned short* __restrict__ ctx)
{
  const int row = blockIdx.x, tid = threadIdx.x;
  const float* hr = h + (size_t)row * 1024;
  float v[4]; float s = 0.f, s2 = 0.f;
  #pragma unroll
  for (int i = 0; i < 4; ++i) {
    const int d = tid + i * 256;
    v[i] = hr[d]; s += v[i]; s2 += v[i] * v[i];
  }
  __shared__ float r1[256], r2[256];
  r1[tid] = s; r2[tid] = s2; __syncthreads();
  for (int o = 128; o > 0; o >>= 1) {
    if (tid < o) { r1[tid] += r1[tid + o]; r2[tid] += r2[tid + o]; }
    __syncthreads();
  }
  const float mu = r1[0] * (1.f / 1024.f);
  const float var = r2[0] * (1.f / 1024.f) - mu * mu;
  const float rstd = rsqrtf(var + 1e-5f);
  #pragma unroll
  for (int i = 0; i < 4; ++i) {
    const int d = tid + i * 256;
    const float x = (v[i] - mu) * rstd * g[d] + bvec[d];
    ctx[(size_t)row * 1024 + d] = f2bf(gelu_f(x));
  }
}

// ---------------------------------------------------------------------------
// One-query attention per (b,h): scores over S=512, softmax, weighted V.
// Reads bf16 K/V (L2-resident); bf16 output feeds the Wo GEMM.
// ---------------------------------------------------------------------------
__global__ __launch_bounds__(256) void attn_kernel(
    const float* __restrict__ qh, const unsigned short* __restrict__ kb,
    const unsigned short* __restrict__ vb, unsigned short* __restrict__ out)
{
  const int b = blockIdx.x >> 3, hh = blockIdx.x & 7, tid = threadIdx.x;
  __shared__ float qs[128];
  __shared__ float sc[512];
  __shared__ float red[256];
  if (tid < 128) qs[tid] = qh[b * 1024 + hh * 128 + tid];
  __syncthreads();
  const float scale = 0.08838834764831845f;  // 1/sqrt(128)
  for (int s = tid; s < 512; s += 256) {
    const unsigned short* kr = kb + ((size_t)(b * 512 + s)) * 1024 + hh * 128;
    float acc = 0.f;
    #pragma unroll 8
    for (int d = 0; d < 128; ++d) acc += qs[d] * bf2f(kr[d]);
    sc[s] = acc * scale;
  }
  __syncthreads();
  red[tid] = fmaxf(sc[tid], sc[tid + 256]);
  __syncthreads();
  for (int o = 128; o > 0; o >>= 1) {
    if (tid < o) red[tid] = fmaxf(red[tid], red[tid + o]);
    __syncthreads();
  }
  const float mx = red[0];
  __syncthreads();
  const float e0 = expf(sc[tid] - mx), e1 = expf(sc[tid + 256] - mx);
  sc[tid] = e0; sc[tid + 256] = e1;
  red[tid] = e0 + e1;
  __syncthreads();
  for (int o = 128; o > 0; o >>= 1) {
    if (tid < o) red[tid] += red[tid + o];
    __syncthreads();
  }
  const float inv = 1.0f / red[0];
  __syncthreads();
  const int d = tid >> 1, part = tid & 1;
  float acc = 0.f;
  const int s0 = part * 256;
  for (int s = s0; s < s0 + 256; ++s)
    acc += sc[s] * bf2f(vb[((size_t)(b * 512 + s)) * 1024 + hh * 128 + d]);
  red[tid] = acc;
  __syncthreads();
  if (part == 0)
    out[b * 1024 + hh * 128 + d] = f2bf((red[2 * d] + red[2 * d + 1]) * inv);
}

// ---------------------------------------------------------------------------
// sync(): causal exp-decay correlation of zbuf pairs; bf16 out feeds GEMMs.
// ---------------------------------------------------------------------------
__global__ void sync_kernel(
    const float* __restrict__ zbuf,
    const float* __restrict__ soft_a, const float* __restrict__ soft_o,
    const int* __restrict__ iia, const int* __restrict__ jja,
    const int* __restrict__ iio, const int* __restrict__ jjo,
    unsigned short* __restrict__ s_act, unsigned short* __restrict__ s_out, int t)
{
  const int idx = blockIdx.x * blockDim.x + threadIdx.x;  // 65536
  const int which = idx >> 15;
  const int r = idx & 32767;
  const int b = r >> 9, s = r & 511;
  const float soft = which ? soft_o[s] : soft_a[s];
  const int ii = which ? iio[s] : iia[s];
  const int jj = which ? jjo[s] : jja[s];
  float num = 0.f, dw = 0.f;
  const float* zb = zbuf + (size_t)b * 17 * 1024;
  for (int tt = 0; tt <= t; ++tt) {
    const float w = expf(-soft * (float)(t - tt));
    const float* zr = zb + tt * 1024;
    num += zr[ii] * zr[jj] * w;
    dw += w * w;
  }
  const unsigned short val = f2bf(num / sqrtf(dw + 1e-8f));
  if (which) s_out[r] = val; else s_act[r] = val;
}

// ---------------------------------------------------------------------------
// per-(b,d) neuron MLP: hist shift+append, 16->4->4->1, write z / zbuf[t+1].
// ---------------------------------------------------------------------------
__global__ void neuron_kernel(
    const float* __restrict__ pre,
    const float* __restrict__ Wn1, const float* __restrict__ bn1,
    const float* __restrict__ Wn2, const float* __restrict__ bn2,
    const float* __restrict__ Wn3, const float* __restrict__ bn3,
    float* __restrict__ hist, float* __restrict__ zv,
    float* __restrict__ zbuf, int t)
{
  const int idx = blockIdx.x * blockDim.x + threadIdx.x;  // 65536
  const int b = idx >> 10, d = idx & 1023;
  float* hp = hist + (size_t)idx * 16;
  float hv[16];
  #pragma unroll
  for (int m = 0; m < 15; ++m) hv[m] = hp[m + 1];
  hv[15] = pre[idx];
  #pragma unroll
  for (int m = 0; m < 16; ++m) hp[m] = hv[m];
  float x1[4];
  #pragma unroll
  for (int h = 0; h < 4; ++h) {
    float a = bn1[d * 4 + h];
    const float* wr = Wn1 + (size_t)d * 64 + h * 16;
    #pragma unroll
    for (int m = 0; m < 16; ++m) a += hv[m] * wr[m];
    x1[h] = gelu_f(a);
  }
  float x2[4];
  #pragma unroll
  for (int g = 0; g < 4; ++g) {
    float a = bn2[d * 4 + g];
    const float* wr = Wn2 + (size_t)d * 16 + g * 4;
    #pragma unroll
    for (int h = 0; h < 4; ++h) a += x1[h] * wr[h];
    x2[g] = gelu_f(a);
  }
  float z = bn3[d];
  #pragma unroll
  for (int h = 0; h < 4; ++h) z += x2[h] * Wn3[d * 4 + h];
  zv[idx] = z;
  zbuf[((size_t)b * 17 + (t + 1)) * 1024 + d] = z;
}

__global__ void softplus_kernel(const float* __restrict__ da, const float* __restrict__ dox,
                                float* __restrict__ sa, float* __restrict__ so)
{
  const int i = blockIdx.x * blockDim.x + threadIdx.x;  // 1024
  if (i < 512) { const float x = da[i];        sa[i]       = (x > 20.f) ? x : log1pf(expf(x)); }
  else         { const float x = dox[i - 512]; so[i - 512] = (x > 20.f) ? x : log1pf(expf(x)); }
}

__global__ void init_kernel(const float* __restrict__ z_init, const float* __restrict__ hist_init,
                            float* __restrict__ zv, float* __restrict__ zbuf,
                            float* __restrict__ hist)
{
  const int idx = blockIdx.x * blockDim.x + threadIdx.x;  // 65536
  const int b = idx >> 10, d = idx & 1023;
  const float zi = z_init[d];
  zv[idx] = zi;
  zbuf[(size_t)b * 17 * 1024 + d] = zi;   // only row 0 is ever read before written
  const float* hs = hist_init + d * 16;
  float* hp = hist + (size_t)idx * 16;
  #pragma unroll
  for (int m = 0; m < 16; ++m) hp[m] = hs[m];
}

__global__ void copyz_kernel(const float* __restrict__ zv, unsigned short* __restrict__ comb)
{
  const int idx = blockIdx.x * blockDim.x + threadIdx.x;  // 65536
  comb[(size_t)(idx >> 10) * 2048 + (idx & 1023)] = f2bf(zv[idx]);
}

// logits[b] = dot(t1[b,:], Wh2) + bh2 ; out layout (B, T, 1)
__global__ void head_kernel(const float* __restrict__ t1, const float* __restrict__ Wh2,
                            const float* __restrict__ bh2, float* __restrict__ out, int t)
{
  __shared__ float red[128];
  const int b = blockIdx.x, tid = threadIdx.x;
  float acc = 0.f;
  for (int i = tid; i < 512; i += 128) acc += t1[b * 512 + i] * Wh2[i];
  red[tid] = acc; __syncthreads();
  for (int o = 64; o > 0; o >>= 1) {
    if (tid < o) red[tid] += red[tid + o];
    __syncthreads();
  }
  if (tid == 0) out[b * 16 + t] = red[0] + bh2[0];
}

// ---------------------------------------------------------------------------
extern "C" void kernel_launch(void* const* d_in, const int* in_sizes, int n_in,
                              void* d_out, int out_size, void* d_ws, size_t ws_size,
                              hipStream_t stream)
{
  const float* emb     = (const float*)d_in[0];
  const float* Wp      = (const float*)d_in[1];
  const float* bp      = (const float*)d_in[2];
  const float* ln_g    = (const float*)d_in[3];
  const float* ln_b    = (const float*)d_in[4];
  const float* decay_a = (const float*)d_in[5];
  const float* decay_o = (const float*)d_in[6];
  const float* Wq      = (const float*)d_in[7];
  const float* bq      = (const float*)d_in[8];
  const float* Wqkv    = (const float*)d_in[9];
  const float* bqkv    = (const float*)d_in[10];
  const float* Wo      = (const float*)d_in[11];
  const float* bo      = (const float*)d_in[12];
  const float* Ws1     = (const float*)d_in[13];
  const float* bs1     = (const float*)d_in[14];
  const float* Ws2     = (const float*)d_in[15];
  const float* bs2     = (const float*)d_in[16];
  const float* Wn1     = (const float*)d_in[17];
  const float* bn1     = (const float*)d_in[18];
  const float* Wn2     = (const float*)d_in[19];
  const float* bn2     = (const float*)d_in[20];
  const float* Wn3     = (const float*)d_in[21];
  const float* bn3     = (const float*)d_in[22];
  const float* Wh1     = (const float*)d_in[23];
  const float* bh1     = (const float*)d_in[24];
  const float* Wh2     = (const float*)d_in[25];
  const float* bh2     = (const float*)d_in[26];
  const float* z_init  = (const float*)d_in[27];
  const float* hist_init = (const float*)d_in[28];
  const int* iia = (const int*)d_in[29];
  const int* jja = (const int*)d_in[30];
  const int* iio = (const int*)d_in[31];
  const int* jjo = (const int*)d_in[32];
  float* out = (float*)d_out;

  // ---- workspace carve-up (K/V reuse the dead h buffer; peak ~270 MB) ----
  char* w = (char*)d_ws;
  size_t off = 0;
  auto alloc = [&](size_t bytes) -> void* {
    void* p = w + off; off += (bytes + 255) & ~(size_t)255; return p;
  };
  const size_t R = 32768;  // B*S rows
  float* h_buf = (float*)alloc(R * 1024 * 4);                  // 134 MB, dead after LN
  unsigned short* ctx = (unsigned short*)alloc(R * 1024 * 2);  // 67 MB bf16
  unsigned short* kb = (unsigned short*)h_buf;                 // bf16 K reuses h
  unsigned short* vb = kb + R * 1024;                          // bf16 V
  // bf16 copies of everything the WMMA GEMMs touch
  unsigned short* emb_bf  = (unsigned short*)alloc(R * 384 * 2);
  unsigned short* Wp_bf   = (unsigned short*)alloc((size_t)1024 * 384 * 2);
  unsigned short* Wq_bf   = (unsigned short*)alloc((size_t)1024 * 512 * 2);
  unsigned short* Wqkv_bf = (unsigned short*)alloc((size_t)3 * 1024 * 1024 * 2);
  unsigned short* Wo_bf   = (unsigned short*)alloc((size_t)1024 * 1024 * 2);
  unsigned short* Ws1_bf  = (unsigned short*)alloc((size_t)2048 * 2048 * 2);
  unsigned short* Ws2_bf  = (unsigned short*)alloc((size_t)1024 * 2048 * 2);
  unsigned short* Wh1_bf  = (unsigned short*)alloc((size_t)512 * 512 * 2);
  float* soft_a = (float*)alloc(512 * 4);
  float* soft_o = (float*)alloc(512 * 4);
  float* zbuf = (float*)alloc((size_t)64 * 17 * 1024 * 4);
  float* zv   = (float*)alloc((size_t)64 * 1024 * 4);
  float* hist = (float*)alloc((size_t)64 * 1024 * 16 * 4);
  unsigned short* s_act = (unsigned short*)alloc((size_t)64 * 512 * 2);
  unsigned short* s_out = (unsigned short*)alloc((size_t)64 * 512 * 2);
  unsigned short* qbuf  = (unsigned short*)alloc((size_t)64 * 1024 * 2);
  float*          qhb   = (float*)alloc((size_t)64 * 1024 * 4);
  unsigned short* aop   = (unsigned short*)alloc((size_t)64 * 1024 * 2);
  unsigned short* comb  = (unsigned short*)alloc((size_t)64 * 2048 * 2);
  unsigned short* s1b   = (unsigned short*)alloc((size_t)64 * 2048 * 2);
  float* preb = (float*)alloc((size_t)64 * 1024 * 4);
  float* t1b  = (float*)alloc((size_t)64 * 512 * 4);

  auto cvt = [&](const float* s, unsigned short* d2, size_t n) {
    const int n4 = (int)(n >> 2);
    cvt_kernel<<<(n4 + 255) / 256, 256, 0, stream>>>(s, d2, n4);
  };

  // ---- setup: bf16 conversions + state init ----
  cvt(emb,  emb_bf,  R * 384);
  cvt(Wp,   Wp_bf,   (size_t)1024 * 384);
  cvt(Wq,   Wq_bf,   (size_t)1024 * 512);
  cvt(Wqkv, Wqkv_bf, (size_t)3 * 1024 * 1024);
  cvt(Wo,   Wo_bf,   (size_t)1024 * 1024);
  cvt(Ws1,  Ws1_bf,  (size_t)2048 * 2048);
  cvt(Ws2,  Ws2_bf,  (size_t)1024 * 2048);
  cvt(Wh1,  Wh1_bf,  (size_t)512 * 512);
  softplus_kernel<<<4, 256, 0, stream>>>(decay_a, decay_o, soft_a, soft_o);
  init_kernel<<<256, 256, 0, stream>>>(z_init, hist_init, zv, zbuf, hist);

  // ---- stage A: big GEMMs ----
  gemm_wmma_kernel<float, false><<<dim3(16, 512), 256, 0, stream>>>(
      emb_bf, 384, Wp_bf, 384, bp, h_buf, 1024, 384);
  ln_gelu_kernel<<<32768, 256, 0, stream>>>(h_buf, ln_g, ln_b, ctx);
  gemm_wmma_kernel<unsigned short, false><<<dim3(16, 512), 256, 0, stream>>>(
      ctx, 1024, Wqkv_bf + (size_t)1024 * 1024, 1024, bqkv + 1024, kb, 1024, 1024);
  gemm_wmma_kernel<unsigned short, false><<<dim3(16, 512), 256, 0, stream>>>(
      ctx, 1024, Wqkv_bf + (size_t)2 * 1024 * 1024, 1024, bqkv + 2048, vb, 1024, 1024);

  // ---- 16-step recurrence ----
  for (int t = 0; t < 16; ++t) {
    sync_kernel<<<256, 256, 0, stream>>>(zbuf, soft_a, soft_o, iia, jja, iio, jjo,
                                         s_act, s_out, t);
    gemm_wmma_kernel<unsigned short, false><<<dim3(16, 1), 256, 0, stream>>>(
        s_act, 512, Wq_bf, 512, bq, qbuf, 1024, 512);
    gemm_wmma_kernel<float, false><<<dim3(16, 1), 256, 0, stream>>>(
        qbuf, 1024, Wqkv_bf, 1024, bqkv, qhb, 1024, 1024);
    attn_kernel<<<512, 256, 0, stream>>>(qhb, kb, vb, aop);
    gemm_wmma_kernel<unsigned short, false><<<dim3(16, 1), 256, 0, stream>>>(
        aop, 1024, Wo_bf, 1024, bo, comb + 1024, 2048, 1024);   // right half of concat
    copyz_kernel<<<256, 256, 0, stream>>>(zv, comb);             // left half = z
    gemm_wmma_kernel<unsigned short, true><<<dim3(32, 1), 256, 0, stream>>>(
        comb, 2048, Ws1_bf, 2048, bs1, s1b, 2048, 2048);
    gemm_wmma_kernel<float, false><<<dim3(16, 1), 256, 0, stream>>>(
        s1b, 2048, Ws2_bf, 2048, bs2, preb, 1024, 2048);
    neuron_kernel<<<256, 256, 0, stream>>>(preb, Wn1, bn1, Wn2, bn2, Wn3, bn3,
                                           hist, zv, zbuf, t);
    gemm_wmma_kernel<float, true><<<dim3(8, 1), 256, 0, stream>>>(
        s_out, 512, Wh1_bf, 512, bh1, t1b, 512, 512);
    head_kernel<<<64, 128, 0, stream>>>(t1b, Wh2, bh2, out, t);
  }
}
